// Block_35923106464322
// MI455X (gfx1250) — matrix operand order repaired and checked by reading, not compiled, not run on hardware
//
#include <hip/hip_runtime.h>
#include <hip/hip_bf16.h>

typedef __attribute__((ext_vector_type(16))) _Float16 v16h;
typedef __attribute__((ext_vector_type(8)))  float    v8f;

#define NROWS   1000000
#define FEATN   128

// ws fragment layout (in dwords). B-frag: [kt][nt][lane][v] packed f16 pairs.
#define LEN_L1  (5*4*32*8)   // K=160 (149 padded), N=64
#define LEN_H   (2*4*32*8)   // K=64, N=64
#define OFF1    0
#define OFF2    (OFF1 + LEN_L1)
#define OFF3    (OFF2 + LEN_H)
#define OFF4    (OFF3 + LEN_H)
#define WSTOT   (OFF4 + LEN_H)

// Per-wave LDS ordering only: CDNA5 LDS (DS) instructions from one wave execute
// in order (ISA 7.3), and all lanes of a wave issue reads after writes in program
// order, so no hardware wait/barrier is needed for wave-private buffers. We only
// have to stop the compiler from reordering LDS accesses across this point.
__device__ inline void wave_lds_fence() {
    __builtin_amdgcn_wave_barrier();
    asm volatile("" ::: "memory");
}

// ---------------- prep: f32 weights -> f16 pre-swizzled B fragments ----------------
__global__ void prep_frags(const float* __restrict__ W1, const float* __restrict__ W2,
                           const float* __restrict__ W3, const float* __restrict__ W4,
                           unsigned int* __restrict__ ws) {
    int t = blockIdx.x * 256 + threadIdx.x;
    if (t >= WSTOT) return;
    const float* W; int Kact, Ncols, off;
    if      (t < OFF2) { W = W1; Kact = 149; Ncols = 64; off = OFF1; }
    else if (t < OFF3) { W = W2; Kact = 64;  Ncols = 64; off = OFF2; }
    else if (t < OFF4) { W = W3; Kact = 64;  Ncols = 64; off = OFF3; }
    else               { W = W4; Kact = 64;  Ncols = 63; off = OFF4; }
    int local = t - off;
    int v    = local & 7;
    int lane = (local >> 3) & 31;
    int nt   = (local >> 8) & 3;
    int kt   = local >> 10;
    int g    = lane >> 4;
    int n    = nt * 16 + (lane & 15);
    int k0   = kt * 32 + ((v >> 2) << 4) + (g << 3) + ((v & 3) << 1);
    float f0 = (k0     < Kact && n < Ncols) ? W[(size_t)k0       * Ncols + n] : 0.f;
    float f1 = (k0 + 1 < Kact && n < Ncols) ? W[(size_t)(k0 + 1) * Ncols + n] : 0.f;
    _Float16 h0 = (_Float16)f0, h1 = (_Float16)f1;
    unsigned int u = ((unsigned int)__builtin_bit_cast(unsigned short, h1) << 16)
                   |  (unsigned int)__builtin_bit_cast(unsigned short, h0);
    ws[t] = u;
}

// ---------------- main fused kernel ----------------
__launch_bounds__(256)
__global__ void flow_kernel(const float* __restrict__ y, const float* __restrict__ logj_in,
                            const float* __restrict__ feat,
                            const float* __restrict__ b1, const float* __restrict__ b2,
                            const float* __restrict__ b3, const float* __restrict__ b4,
                            const unsigned int* __restrict__ frags,
                            float* __restrict__ out) {
    __shared__ float sbuf[8][16][68];   // per-wave activation / st tile (stride 68: conflict-free)
    __shared__ float ebuf[8][16][24];   // per-wave positional embeddings

    const int lane = threadIdx.x & 31;
    const int wv   = threadIdx.x >> 5;
    const int g    = lane >> 4;
    const int mrow = lane & 15;
    const int rowBase = blockIdx.x * 128 + wv * 16;
    int rowg = rowBase + mrow; if (rowg > NROWS - 1) rowg = NROWS - 1;

    float (*sw)[68] = sbuf[wv];
    float (*ew)[24] = ebuf[wv];

    // ---- positional embedding: lanes 0..15 each handle one row ----
    if (lane < 16) {
        const float* yr = y + (size_t)rowg * 6;
        float x0 = yr[0], x1 = yr[1], x2 = yr[2];
        ew[lane][0] = x0; ew[lane][1] = x1; ew[lane][2] = x2;
        const float fr[3] = {1.f, 2.f, 4.f};
        float xs[3] = {x0, x1, x2};
        #pragma unroll
        for (int fi = 0; fi < 3; ++fi) {
            #pragma unroll
            for (int c = 0; c < 3; ++c) {
                float ang = xs[c] * fr[fi];
                ew[lane][3 + 6 * fi + c] = __sinf(ang);
                ew[lane][6 + 6 * fi + c] = __cosf(ang);
            }
        }
    }
    wave_lds_fence();

    const float* featRow = feat + (size_t)rowg * FEATN;
    const float* erow    = ew[mrow];

    // ---- layer 1: [16x160] x [160x64], A built on the fly, h = 2*in - 1 ----
    v8f acc[4];
    #pragma unroll
    for (int i = 0; i < 4; ++i) acc[i] = v8f{0.f,0.f,0.f,0.f,0.f,0.f,0.f,0.f};

    #pragma unroll
    for (int kt = 0; kt < 5; ++kt) {
        v16h a;
        #pragma unroll
        for (int v = 0; v < 8; ++v) {
            int K0 = kt * 32 + ((v >> 2) << 4) + (g << 3) + ((v & 3) << 1);
            int K1 = K0 + 1;
            float f0 = (K0 < 21) ? erow[K0] : ((K0 < 149) ? featRow[K0 - 21] : 0.f);
            float f1 = (K1 < 21) ? erow[K1] : ((K1 < 149) ? featRow[K1 - 21] : 0.f);
            a[2 * v]     = (_Float16)(2.f * f0 - 1.f);
            a[2 * v + 1] = (_Float16)(2.f * f1 - 1.f);
        }
        #pragma unroll
        for (int nt = 0; nt < 4; ++nt) {
            const v16h* bp = (const v16h*)(frags + OFF1 + ((size_t)((kt * 4 + nt) * 32 + lane)) * 8);
            acc[nt] = __builtin_amdgcn_wmma_f32_16x16x32_f16(false, a, false, *bp,
                                                             (short)0, acc[nt], false, false);
        }
    }
    // bias + lrelu, write h1 to LDS
    #pragma unroll
    for (int nt = 0; nt < 4; ++nt)
        #pragma unroll
        for (int r = 0; r < 8; ++r) {
            int M = r + g * 8, Nn = nt * 16 + mrow;
            float vv = acc[nt][r] + b1[Nn];
            vv = (vv >= 0.f) ? vv : 0.01f * vv;
            sw[M][Nn] = vv;
        }
    wave_lds_fence();

    // ---- layers 2..4: [16x64] x [64x64] via wave-private LDS handoff ----
    auto layer64 = [&](const unsigned int* wf, const float* bias, int bN, bool relu) {
        v8f ac[4];
        #pragma unroll
        for (int i = 0; i < 4; ++i) ac[i] = v8f{0.f,0.f,0.f,0.f,0.f,0.f,0.f,0.f};
        #pragma unroll
        for (int kt = 0; kt < 2; ++kt) {
            v16h a;
            #pragma unroll
            for (int v = 0; v < 8; ++v) {
                int K0 = kt * 32 + ((v >> 2) << 4) + (g << 3) + ((v & 3) << 1);
                a[2 * v]     = (_Float16)sw[mrow][K0];
                a[2 * v + 1] = (_Float16)sw[mrow][K0 + 1];
            }
            #pragma unroll
            for (int nt = 0; nt < 4; ++nt) {
                const v16h* bp = (const v16h*)(wf + ((size_t)(((kt << 2) + nt) * 32 + lane)) * 8);
                ac[nt] = __builtin_amdgcn_wmma_f32_16x16x32_f16(false, a, false, *bp,
                                                                (short)0, ac[nt], false, false);
            }
        }
        wave_lds_fence();   // all A reads of sbuf ordered before overwrite
        #pragma unroll
        for (int nt = 0; nt < 4; ++nt)
            #pragma unroll
            for (int r = 0; r < 8; ++r) {
                int M = r + g * 8, Nn = nt * 16 + mrow;
                float vv = ac[nt][r] + ((Nn < bN) ? bias[Nn] : 0.f);
                if (relu) vv = (vv >= 0.f) ? vv : 0.01f * vv;
                sw[M][Nn] = vv;
            }
        wave_lds_fence();
    };

    layer64(frags + OFF2, b2, 64, true);
    layer64(frags + OFF3, b3, 64, true);
    layer64(frags + OFF4, b4, 63, false);   // st tile now in sbuf: [row][k*21 + j]

    // ---- piecewise-quadratic CDF inversion: 48 (row,k) items over 32 lanes ----
    auto pwq = [&](int row, int k) -> float {
        int rg = rowBase + row;
        bool valid = rg < NROWS;
        if (rg > NROWS - 1) rg = NROWS - 1;
        float yv = y[(size_t)rg * 6 + 3 + k];
        const float* st = &sw[row][k * 21];
        const float eps = 1.1920929e-07f;

        float w[10], wc[10];
        float cs = 0.f;
        #pragma unroll
        for (int j = 0; j < 10; ++j) { float e = __expf(st[11 + j]); cs += e; w[j] = e; wc[j] = cs; }
        float inv = 1.f / cs;
        #pragma unroll
        for (int j = 0; j < 10; ++j) { w[j] *= inv; wc[j] *= inv; }

        float ev[11];
        #pragma unroll
        for (int j = 0; j < 11; ++j) ev[j] = __expf(st[j]);
        float den = 0.f;
        #pragma unroll
        for (int j = 0; j < 10; ++j) den += (ev[j] + ev[j + 1]) * 0.5f * w[j];
        float dinv = 1.f / den;
        float vv[11];
        #pragma unroll
        for (int j = 0; j < 11; ++j) vv[j] = fmaxf(ev[j] * dinv, 1e-6f);

        float vwa[11];
        vwa[0] = 0.f;
        #pragma unroll
        for (int i = 1; i <= 10; ++i) vwa[i] = vwa[i - 1] + (vv[i - 1] + vv[i]) * 0.5f * w[i - 1];

        // select largest bin edge e in [0,9] with vwa[e] <= y
        float ve = vv[0], ve1 = vv[1], we = w[0], vwe = 0.f, wse = 0.f;
        #pragma unroll
        for (int i = 1; i <= 9; ++i) {
            bool c = vwa[i] <= yv;
            if (c) { ve = vv[i]; ve1 = vv[i + 1]; we = w[i]; vwe = vwa[i]; wse = wc[i - 1]; }
        }

        float a  = (ve1 - ve) * we;
        float bb = ve * we;
        float c2 = vwe - yv;
        if (fabsf(a) < eps) a = eps;
        float d  = fmaxf(bb * bb - 2.f * a * c2, 0.f);
        float sq = sqrtf(d);
        float s1 = (-bb - sq) / a, s2 = (-bb + sq) / a;
        float sol = (s1 >= 0.f && s1 < 1.f) ? s1 : s2;
        sol = fminf(fmaxf(sol, eps), 1.f - eps);
        float xm = we * sol + wse;
        xm = fminf(fmaxf(xm, eps), 1.f - eps);
        float lj = -__logf(ve + (ve1 - ve) * sol);
        if (valid) out[(size_t)rg * 6 + 3 + k] = xm;
        return lj;
    };

    float lj01 = pwq(mrow, g);                 // lanes 0-15: k=0 ; lanes 16-31: k=1
    float lj2  = 0.f;
    if (lane < 16) lj2 = pwq(mrow, 2);         // lanes 0-15 also: k=2
    float lj1 = __shfl(lj01, lane + 16, 32);   // lanes<16 fetch k=1 term from partner lane

    // ---- finalize: copy y_n, reduce logj (no LDS, pure wave shuffle) ----
    if (lane < 16) {
        int rg = rowBase + mrow;
        if (rg < NROWS) {
            out[(size_t)6 * NROWS + rg] = logj_in[rg] + lj01 + lj1 + lj2;
            const float* yr = y + (size_t)rg * 6;
            out[(size_t)rg * 6 + 0] = yr[0];
            out[(size_t)rg * 6 + 1] = yr[1];
            out[(size_t)rg * 6 + 2] = yr[2];
        }
    }
}

extern "C" void kernel_launch(void* const* d_in, const int* in_sizes, int n_in,
                              void* d_out, int out_size, void* d_ws, size_t ws_size,
                              hipStream_t stream) {
    const float* y    = (const float*)d_in[0];
    const float* logj = (const float*)d_in[1];
    const float* feat = (const float*)d_in[2];
    const float* W1   = (const float*)d_in[3];
    const float* b1   = (const float*)d_in[4];
    const float* W2   = (const float*)d_in[5];
    const float* b2   = (const float*)d_in[6];
    const float* W3   = (const float*)d_in[7];
    const float* b3   = (const float*)d_in[8];
    const float* W4   = (const float*)d_in[9];
    const float* b4   = (const float*)d_in[10];
    unsigned int* frags = (unsigned int*)d_ws;

    prep_frags<<<(WSTOT + 255) / 256, 256, 0, stream>>>(W1, W2, W3, W4, frags);

    int blocks = (NROWS + 127) / 128;   // 128 rows per block (8 waves x 16-row tiles)
    flow_kernel<<<blocks, 256, 0, stream>>>(y, logj, feat, b1, b2, b3, b4, frags,
                                            (float*)d_out);
}
